// UEDDIENetwork_34136400068758
// MI455X (gfx1250) — matrix operand order repaired and checked by reading, not compiled, not run on hardware
//
#include <hip/hip_runtime.h>

typedef __attribute__((ext_vector_type(16))) _Float16 v16h;
typedef __attribute__((ext_vector_type(8)))  float    v8f;
typedef __attribute__((ext_vector_type(4)))  unsigned int u32x4;
typedef __attribute__((ext_vector_type(8)))  int          i32x8;
typedef __attribute__((ext_vector_type(4)))  int          i32x4;

#define D_    128
#define H_    128
#define N_    512
#define NEXP_ 4
#define TPB_TILES 8

#if defined(__has_builtin)
#if __has_builtin(__builtin_amdgcn_tensor_load_to_lds)
#define USE_TDM 1
#endif
#endif

__global__ void zero_out_kernel(float* __restrict__ out, int n) {
    int i = blockIdx.x * blockDim.x + threadIdx.x;
    if (i < n) out[i] = 0.0f;
}

#ifdef USE_TDM
// Issue a TDM load of a 16x128 f32 tile (row-major, stride 128) into LDS.
// Descriptor per CDNA5 ISA (2D, groups 2/3 zero).
// This toolchain exposes the 6-arg builtin:
//   (u32x4 g0, i32x8 g1, i32x4 g2, i32x4 g3, i32x8 pad, i32 cpol)
__device__ inline void tdm_load_tile(const float* gptr, unsigned lds_off) {
    unsigned long long ga = (unsigned long long)(__UINTPTR_TYPE__)gptr;
    u32x4 g0;
    g0[0] = 1u;                                   // count=1 (valid user D#)
    g0[1] = lds_off;                              // lds_addr (bytes)
    g0[2] = (unsigned)(ga & 0xffffffffu);         // global_addr[31:0]
    g0[3] = (unsigned)((ga >> 32) & 0x01ffffffu)  // global_addr[56:32]
          | 0x80000000u;                          // type = 2 ("image")
    i32x8 g1;
    g1[0] = 0x00020000;          // workgroup_mask=0, data_size=2 (4B)
    g1[1] = (int)(128u << 16);   // tensor_dim0 = 128 (elements)
    g1[2] = (int)(16u  << 16);   // tensor_dim1 = 16
    g1[3] = (int)(128u << 16);   // tile_dim0   = 128
    g1[4] = 16;                  // tile_dim1   = 16, tile_dim2 = 0
    g1[5] = 128;                 // tensor_dim0_stride = 128 (elements)
    g1[6] = 0;                   // stride hi / tensor_dim1_stride lo
    g1[7] = 0;
    i32x4 z4 = {0, 0, 0, 0};
    i32x8 z8 = {0, 0, 0, 0, 0, 0, 0, 0};
    __builtin_amdgcn_tensor_load_to_lds(g0, g1, z4, z4, z8, 0);
}
#endif

__global__ __launch_bounds__(256)
void moe_wmma_kernel(const float* __restrict__ X,
                     const long long* __restrict__ E,
                     const float* __restrict__ W1,
                     const float* __restrict__ b1,
                     const float* __restrict__ W2,
                     const float* __restrict__ b2,
                     float* __restrict__ out,
                     int ntiles)
{
    __shared__ float Xbuf[2][16 * D_];    // 2 x 8 KB f32 token tiles (ping-pong)
    __shared__ float acc[16 * NEXP_];     // per (token, expert) scalar

    const int tid  = threadIdx.x;
    const int lane = tid & 31;
    const int wid  = tid >> 5;            // 0..7 (8 waves)
    const int e    = wid >> 1;            // expert id 0..3
    const int h0b  = (wid & 1) * 64;      // which half of H this wave owns
    const int col  = lane & 15;

    // ---------------- one-time: W1[e] -> f16 WMMA-B fragments in VGPRs ------
    // B (K=32 x N=16): lane n (0-15) holds K=0..15 of column n in 16 halves;
    // lane n+16 holds K=16..31 of column n.
    const int kbB = (lane < 16) ? 0 : 16;
    v16h bfrag[4][4];
    #pragma unroll
    for (int ks = 0; ks < 4; ++ks) {
        #pragma unroll
        for (int ct = 0; ct < 4; ++ct) {
            const int h = h0b + ct * 16 + col;
            const float* wp = W1 + ((size_t)e * D_ + (ks * 32 + kbB)) * H_ + h;
            v16h f;
            #pragma unroll
            for (int j = 0; j < 16; ++j) f[j] = (_Float16)wp[j * H_];
            // Pin the converted f16 fragment in registers: prevents the
            // compiler from keeping f32 sources live and re-converting
            // inside the tile loop (observed remat blowing past 256 VGPRs).
            asm volatile("" : "+v"(f));
            bfrag[ks][ct] = f;
        }
    }
    float b1v[4], w2v[4];
    #pragma unroll
    for (int ct = 0; ct < 4; ++ct) {
        const int h = h0b + ct * 16 + col;
        b1v[ct] = b1[e * H_ + h];
        w2v[ct] = W2[e * H_ + h];
    }

    const int tiles_per_row = N_ / 16;
    const int tile0 = blockIdx.x * TPB_TILES;

#ifdef USE_TDM
    // Prologue: async-load the first tile, wait, publish.
    if (wid == 0) {
        if (tile0 < ntiles) {
            const int b  = tile0 / tiles_per_row;
            const int n0 = (tile0 % tiles_per_row) * 16;
            tdm_load_tile(X + ((size_t)b * N_ + n0) * D_,
                          (unsigned)(__UINTPTR_TYPE__)&Xbuf[0][0]);
        }
        __builtin_amdgcn_s_wait_tensorcnt(0);
    }
    __syncthreads();
#endif

    #pragma unroll 1
    for (int t = 0; t < TPB_TILES; ++t) {
        const int tile = tile0 + t;
        if (tile >= ntiles) break;
        const int b  = tile / tiles_per_row;
        const int n0 = (tile % tiles_per_row) * 16;
        const float* cur = &Xbuf[t & 1][0];

#ifdef USE_TDM
        // Kick off the NEXT tile's DMA into the other buffer (one wave only;
        // per-wave branch => only wave 0's instruction stream issues the TDM).
        if (wid == 0 && (t + 1) < TPB_TILES && (tile + 1) < ntiles) {
            const int bb = (tile + 1) / tiles_per_row;
            const int nn = ((tile + 1) % tiles_per_row) * 16;
            tdm_load_tile(X + ((size_t)bb * N_ + nn) * D_,
                          (unsigned)(__UINTPTR_TYPE__)&Xbuf[(t + 1) & 1][0]);
        }
#else
        // Fallback: cooperative synchronous load of tile t.
        {
            const float4* Xt = (const float4*)(X + ((size_t)b * N_ + n0) * D_);
            float4* Lb = (float4*)cur;
            Lb[tid]       = Xt[tid];
            Lb[tid + 256] = Xt[tid + 256];
        }
#endif
        if (tid < 16 * NEXP_) acc[tid] = 0.0f;
        __syncthreads();   // acc zeroed (and fallback tile loaded) before use

        // ---- GEMM: 16 tokens x 128 D x 64 H-cols for this wave ------------
        v8f c0 = {}, c1 = {}, c2 = {}, c3 = {};
        const int row = lane & 15;
        const int kbA = (lane < 16) ? 0 : 16;
        #pragma unroll
        for (int ks = 0; ks < 4; ++ks) {
            // A (16x32 f16): lane holds row=lane%16, K = base+{0..15}
            const float* ap = cur + row * D_ + ks * 32 + kbA;
            float4 f0 = *(const float4*)(ap);        // ds_load_b128
            float4 f1 = *(const float4*)(ap + 4);
            float4 f2 = *(const float4*)(ap + 8);
            float4 f3 = *(const float4*)(ap + 12);
            v16h a;
            a[0]  = (_Float16)f0.x; a[1]  = (_Float16)f0.y;
            a[2]  = (_Float16)f0.z; a[3]  = (_Float16)f0.w;
            a[4]  = (_Float16)f1.x; a[5]  = (_Float16)f1.y;
            a[6]  = (_Float16)f1.z; a[7]  = (_Float16)f1.w;
            a[8]  = (_Float16)f2.x; a[9]  = (_Float16)f2.y;
            a[10] = (_Float16)f2.z; a[11] = (_Float16)f2.w;
            a[12] = (_Float16)f3.x; a[13] = (_Float16)f3.y;
            a[14] = (_Float16)f3.z; a[15] = (_Float16)f3.w;
            c0 = __builtin_amdgcn_wmma_f32_16x16x32_f16(false, a, false, bfrag[ks][0], (short)0, c0, false, false);
            c1 = __builtin_amdgcn_wmma_f32_16x16x32_f16(false, a, false, bfrag[ks][1], (short)0, c1, false, false);
            c2 = __builtin_amdgcn_wmma_f32_16x16x32_f16(false, a, false, bfrag[ks][2], (short)0, c2, false, false);
            c3 = __builtin_amdgcn_wmma_f32_16x16x32_f16(false, a, false, bfrag[ks][3], (short)0, c3, false, false);
        }

        // ---- epilogue: softplus + W2 dot, reduce over h in-registers ------
        // C layout: lane holds h-column n = lane%16, token rows
        // m = r + (lane<16 ? 0 : 8). Destination acc index is ct-independent.
        const int mbase = (lane < 16) ? 0 : 8;
        v8f cc[4] = { c0, c1, c2, c3 };
        float part[8];
        #pragma unroll
        for (int r = 0; r < 8; ++r) part[r] = 0.0f;
        #pragma unroll
        for (int ct = 0; ct < 4; ++ct) {
            #pragma unroll
            for (int r = 0; r < 8; ++r) {
                float x  = cc[ct][r] + b1v[ct];
                float sp = (x > 20.0f) ? x : __logf(1.0f + __expf(x));
                part[r] += sp * w2v[ct];
            }
        }
        // butterfly-reduce across the 16 lanes sharing each token set
        #pragma unroll
        for (int off = 1; off <= 8; off <<= 1) {
            #pragma unroll
            for (int r = 0; r < 8; ++r)
                part[r] += __shfl_xor(part[r], off, 32);
        }
        if (col == 0) {
            #pragma unroll
            for (int r = 0; r < 8; ++r)
                atomicAdd(&acc[(mbase + r) * NEXP_ + e], part[r]);
        }
        __syncthreads();   // acc complete; all reads of `cur` finished

        // ---- route by E, reduce over 16 tokens, accumulate -sum -----------
        if (wid == 0) {
            float v = 0.0f;
            if (lane < 16) {
                const int es = (int)E[(size_t)b * N_ + n0 + lane];
                v = acc[lane * NEXP_ + es] + b2[es];
            }
            #pragma unroll
            for (int off = 16; off >= 1; off >>= 1)
                v += __shfl_down(v, off, 32);
            if (lane == 0) atomicAdd(&out[b], -v);
#ifdef USE_TDM
            __builtin_amdgcn_s_wait_tensorcnt(0);  // next tile's DMA landed
#endif
        }
        __syncthreads();   // next buffer visible; acc reads done
    }
}

extern "C" void kernel_launch(void* const* d_in, const int* in_sizes, int n_in,
                              void* d_out, int out_size, void* d_ws, size_t ws_size,
                              hipStream_t stream) {
    const float*     X  = (const float*)d_in[0];
    const long long* E  = (const long long*)d_in[1];
    const float*     W1 = (const float*)d_in[2];
    const float*     b1 = (const float*)d_in[3];
    const float*     W2 = (const float*)d_in[4];
    const float*     b2 = (const float*)d_in[5];
    float* out = (float*)d_out;

    const int tokens = in_sizes[1];        // B*N = 131072
    const int ntiles = tokens / 16;        // 8192 token tiles
    const int grid   = (ntiles + TPB_TILES - 1) / TPB_TILES;

    zero_out_kernel<<<(out_size + 255) / 256, 256, 0, stream>>>(out, out_size);
    moe_wmma_kernel<<<grid, 256, 0, stream>>>(X, E, W1, b1, W2, b2, out, ntiles);
}
